// DecoderPp_85298050498941
// MI455X (gfx1250) — matrix operand order repaired and checked by reading, not compiled, not run on hardware
//
#include <hip/hip_runtime.h>
#include <hip/hip_bf16.h>
#include <math.h>

// ---------------------------------------------------------------------------
// Types for WMMA operands (CDNA5 / gfx1250, wave32)
// ---------------------------------------------------------------------------
typedef __attribute__((ext_vector_type(2)))  float    v2f;
typedef __attribute__((ext_vector_type(8)))  float    v8f;
typedef __attribute__((ext_vector_type(16))) _Float16 v16h;

// ---------------------------------------------------------------------------
// kNN inverse-distance interpolation.
//   One block per (target-chunk, graph). Source positions staged in LDS.
//   Each thread owns one target point, keeps a k-best list in registers,
//   then gathers k source feature rows (L2-resident) and writes the
//   weighted average into the concat buffer (cols [0,C) of row stride ldo).
// ---------------------------------------------------------------------------
template <int KNN>
__global__ __launch_bounds__(256) void knn_interp_kernel(
    const float* __restrict__ srcF,   // [B*ns, C]
    int C,
    const float* __restrict__ srcP,   // [B*ns, 3]
    const float* __restrict__ tgtP,   // [B*nt, 3]
    float* __restrict__ out,          // [B*nt, ldo], writes cols [0,C)
    int ldo, int ns, int nt)
{
    __shared__ float sp[3 * 1024];    // max ns per graph = 1024

    const int g   = blockIdx.y;
    const int tid = threadIdx.x;

    for (int i = tid; i < ns * 3; i += blockDim.x)
        sp[i] = srcP[(size_t)g * ns * 3 + i];
    __syncthreads();

    const int t = blockIdx.x * blockDim.x + tid;   // target index in graph
    const size_t trow = (size_t)g * nt + t;
    const float qx = tgtP[trow * 3 + 0];
    const float qy = tgtP[trow * 3 + 1];
    const float qz = tgtP[trow * 3 + 2];

    float bd[KNN];
    int   bi[KNN];
#pragma unroll
    for (int i = 0; i < KNN; i++) { bd[i] = 3.4e38f; bi[i] = 0; }

    for (int s = 0; s < ns; s++) {
        const float dx = qx - sp[s * 3 + 0];
        const float dy = qy - sp[s * 3 + 1];
        const float dz = qz - sp[s * 3 + 2];
        const float d2 = dx * dx + dy * dy + dz * dz;
        if (d2 < bd[KNN - 1]) {
            bd[KNN - 1] = d2; bi[KNN - 1] = s;
#pragma unroll
            for (int j = KNN - 1; j > 0; j--) {
                if (bd[j] < bd[j - 1]) {
                    float td = bd[j]; bd[j] = bd[j - 1]; bd[j - 1] = td;
                    int   ti = bi[j]; bi[j] = bi[j - 1]; bi[j - 1] = ti;
                }
            }
        }
    }

    float w[KNN];
    float sumw = 0.f;
#pragma unroll
    for (int i = 0; i < KNN; i++) {
        w[i] = 1.0f / fmaxf(bd[i], 1e-16f);
        sumw += w[i];
    }
    const float inv = 1.0f / sumw;

    const float* base = srcF + (size_t)g * ns * C;
    float* orow = out + trow * ldo;
    for (int c = 0; c < C; c++) {
        float acc = 0.f;
#pragma unroll
        for (int i = 0; i < KNN; i++)
            acc += w[i] * base[(size_t)bi[i] * C + c];
        orow[c] = acc * inv;
    }
}

// ---------------------------------------------------------------------------
// Copy skip features into the concat buffer columns [coff, coff+cs).
// ---------------------------------------------------------------------------
__global__ __launch_bounds__(256) void copy_cols_kernel(
    const float* __restrict__ src, int cs,
    float* __restrict__ dst, int ldo, int coff, long n)
{
    long i = (long)blockIdx.x * blockDim.x + threadIdx.x;
    if (i >= n) return;
    long r = i / cs;
    int  c = (int)(i - r * cs);
    dst[r * (long)ldo + coff + c] = src[i];
}

// ---------------------------------------------------------------------------
// WMMA GEMM:  Y[M,N] = act( X[M,K] @ W[N,K]^T + bias[N] )
//
//   fp32 WMMA (V_WMMA_F32_16X16X4_F32): reference precision, engages the
//   CDNA5 matrix pipes. One wave computes a 32x64 macro-tile:
//     2 M-tiles x 4 N-tiles -> 8 independent v8f accumulators (64 VGPRs).
//   Per 4-wide K step: the 2 A + 4 B fragments are loaded into DISTINCT
//   live registers first (so the backend can cluster the 6 global_load_b64s
//   and cover them with one s_wait_loadcnt), then 8 independent WMMAs are
//   issued back-to-back. B row pointers are hoisted out of the K loop.
//   Main K loop is unguarded; the K%4 remainder (K=67 layer) is one guarded
//   tail step.
//
//   ISA operand layouts (wave32):
//     A 16x4  f32: M = lane%16, K = (lane/16)*2 + {0,1}
//     B 4x16  f32: N = lane%16, K = (lane/16)*2 + {0,1}
//     C/D 16x16 f32: col = lane%16, row = r + (lane/16)*8, r = 0..7
//
//   Requires M % 32 == 0 and N % 64 == 0 (all layers satisfy this), so
//   there is no divergence anywhere: EXEC is all ones as WMMA requires.
// ---------------------------------------------------------------------------
__global__ __launch_bounds__(32) void gemm_wmma_kernel(
    const float* __restrict__ X,     // [M,K]
    const float* __restrict__ W,     // [N,K]
    const float* __restrict__ bias,  // [N]
    float* __restrict__ Y,           // [M,N]
    int M, int N, int K, int doTanh)
{
    const int lane = threadIdx.x;      // 0..31
    const int half = lane >> 4;        // 0 or 1
    const int l16  = lane & 15;
    const int m0   = blockIdx.y << 5;  // 32 rows per wave
    const int n0   = blockIdx.x << 6;  // 64 cols per wave

    const float* xr0 = X + (size_t)(m0 + l16) * K;        // A tile 0
    const float* xr1 = xr0 + (size_t)16 * K;              // A tile 1
    const float* wp0 = W + (size_t)(n0 + l16) * K;        // B tiles, hoisted
    const float* wp1 = wp0 + (size_t)16 * K;
    const float* wp2 = wp0 + (size_t)32 * K;
    const float* wp3 = wp0 + (size_t)48 * K;

    v8f c[2][4];
#pragma unroll
    for (int mt = 0; mt < 2; mt++)
#pragma unroll
        for (int nt = 0; nt < 4; nt++)
            c[mt][nt] = (v8f){};

    const int Kmain = K & ~3;

#if __has_builtin(__builtin_amdgcn_wmma_f32_16x16x4_f32)
    for (int kk = 0; kk < Kmain; kk += 4) {
        const int k0 = kk + half * 2;
        // Batch all 6 fragment loads into distinct registers first.
        v2f a0, a1, b0, b1, b2, b3;
        a0[0] = xr0[k0]; a0[1] = xr0[k0 + 1];
        a1[0] = xr1[k0]; a1[1] = xr1[k0 + 1];
        b0[0] = wp0[k0]; b0[1] = wp0[k0 + 1];
        b1[0] = wp1[k0]; b1[1] = wp1[k0 + 1];
        b2[0] = wp2[k0]; b2[1] = wp2[k0 + 1];
        b3[0] = wp3[k0]; b3[1] = wp3[k0 + 1];
        // 8 independent WMMAs (no D->A/B hazards).
        c[0][0] = __builtin_amdgcn_wmma_f32_16x16x4_f32(false, a0, false, b0, (short)0, c[0][0], false, false);
        c[1][0] = __builtin_amdgcn_wmma_f32_16x16x4_f32(false, a1, false, b0, (short)0, c[1][0], false, false);
        c[0][1] = __builtin_amdgcn_wmma_f32_16x16x4_f32(false, a0, false, b1, (short)0, c[0][1], false, false);
        c[1][1] = __builtin_amdgcn_wmma_f32_16x16x4_f32(false, a1, false, b1, (short)0, c[1][1], false, false);
        c[0][2] = __builtin_amdgcn_wmma_f32_16x16x4_f32(false, a0, false, b2, (short)0, c[0][2], false, false);
        c[1][2] = __builtin_amdgcn_wmma_f32_16x16x4_f32(false, a1, false, b2, (short)0, c[1][2], false, false);
        c[0][3] = __builtin_amdgcn_wmma_f32_16x16x4_f32(false, a0, false, b3, (short)0, c[0][3], false, false);
        c[1][3] = __builtin_amdgcn_wmma_f32_16x16x4_f32(false, a1, false, b3, (short)0, c[1][3], false, false);
    }
    if (Kmain < K) {   // single guarded tail step (K = 67 layer only)
        const int k0 = Kmain + half * 2;
        const bool g0 = (k0 < K), g1 = (k0 + 1 < K);
        v2f a0, a1, b0, b1, b2, b3;
        a0[0] = g0 ? xr0[k0] : 0.f;  a0[1] = g1 ? xr0[k0 + 1] : 0.f;
        a1[0] = g0 ? xr1[k0] : 0.f;  a1[1] = g1 ? xr1[k0 + 1] : 0.f;
        b0[0] = g0 ? wp0[k0] : 0.f;  b0[1] = g1 ? wp0[k0 + 1] : 0.f;
        b1[0] = g0 ? wp1[k0] : 0.f;  b1[1] = g1 ? wp1[k0 + 1] : 0.f;
        b2[0] = g0 ? wp2[k0] : 0.f;  b2[1] = g1 ? wp2[k0 + 1] : 0.f;
        b3[0] = g0 ? wp3[k0] : 0.f;  b3[1] = g1 ? wp3[k0 + 1] : 0.f;
        c[0][0] = __builtin_amdgcn_wmma_f32_16x16x4_f32(false, a0, false, b0, (short)0, c[0][0], false, false);
        c[1][0] = __builtin_amdgcn_wmma_f32_16x16x4_f32(false, a1, false, b0, (short)0, c[1][0], false, false);
        c[0][1] = __builtin_amdgcn_wmma_f32_16x16x4_f32(false, a0, false, b1, (short)0, c[0][1], false, false);
        c[1][1] = __builtin_amdgcn_wmma_f32_16x16x4_f32(false, a1, false, b1, (short)0, c[1][1], false, false);
        c[0][2] = __builtin_amdgcn_wmma_f32_16x16x4_f32(false, a0, false, b2, (short)0, c[0][2], false, false);
        c[1][2] = __builtin_amdgcn_wmma_f32_16x16x4_f32(false, a1, false, b2, (short)0, c[1][2], false, false);
        c[0][3] = __builtin_amdgcn_wmma_f32_16x16x4_f32(false, a0, false, b3, (short)0, c[0][3], false, false);
        c[1][3] = __builtin_amdgcn_wmma_f32_16x16x4_f32(false, a1, false, b3, (short)0, c[1][3], false, false);
    }
#else
    // Fallback: fp16 inputs, fp32 accumulate (codegen-confirmed builtin).
    for (int kk = 0; kk < K; kk += 32) {
        const float* wps[4] = {wp0, wp1, wp2, wp3};
        v16h a0, a1;
#pragma unroll
        for (int j = 0; j < 8; j++) {
            const int ka = kk + half * 8 + j;
            a0[j]     = (ka      < K) ? (_Float16)xr0[ka]      : (_Float16)0.f;
            a0[j + 8] = (ka + 16 < K) ? (_Float16)xr0[ka + 16] : (_Float16)0.f;
            a1[j]     = (ka      < K) ? (_Float16)xr1[ka]      : (_Float16)0.f;
            a1[j + 8] = (ka + 16 < K) ? (_Float16)xr1[ka + 16] : (_Float16)0.f;
        }
#pragma unroll
        for (int nt = 0; nt < 4; nt++) {
            v16h b;
#pragma unroll
            for (int j = 0; j < 16; j++) {
                const int kb = kk + half * 8 + (j & 7) + ((j >> 3) << 4);
                b[j] = (kb < K) ? (_Float16)wps[nt][kb] : (_Float16)0.f;
            }
            c[0][nt] = __builtin_amdgcn_wmma_f32_16x16x32_f16(
                           false, a0, false, b, (short)0, c[0][nt], false, false);
            c[1][nt] = __builtin_amdgcn_wmma_f32_16x16x32_f16(
                           false, a1, false, b, (short)0, c[1][nt], false, false);
        }
    }
#endif

    // Epilogue: bias + optional tanh, fused.
#pragma unroll
    for (int nt = 0; nt < 4; nt++) {
        const int col = n0 + nt * 16 + l16;
        const float bv = bias[col];
#pragma unroll
        for (int mt = 0; mt < 2; mt++) {
            const int rbase = m0 + mt * 16 + half * 8;
#pragma unroll
            for (int r = 0; r < 8; r++) {
                float y = c[mt][nt][r] + bv;
                if (doTanh) y = tanhf(y);
                Y[(size_t)(rbase + r) * N + col] = y;
            }
        }
    }
}

// ---------------------------------------------------------------------------
// Final 64 -> 3 linear layer (too narrow for a 16-wide WMMA tile; trivial).
// ---------------------------------------------------------------------------
__global__ __launch_bounds__(256) void final_linear_kernel(
    const float* __restrict__ H,     // [M,64]
    const float* __restrict__ W,     // [3,64]
    const float* __restrict__ b,     // [3]
    float* __restrict__ out,         // [M,3]
    int M)
{
    const int i = blockIdx.x * blockDim.x + threadIdx.x;
    if (i >= M) return;
    const float* h = H + (size_t)i * 64;
#pragma unroll
    for (int c = 0; c < 3; c++) {
        float acc = b[c];
        const float* w = W + c * 64;
#pragma unroll
        for (int k = 0; k < 64; k++) acc += h[k] * w[k];
        out[(size_t)i * 3 + c] = acc;
    }
}

// ---------------------------------------------------------------------------
// Host orchestration
// ---------------------------------------------------------------------------
extern "C" void kernel_launch(void* const* d_in, const int* in_sizes, int n_in,
                              void* d_out, int out_size, void* d_ws, size_t ws_size,
                              hipStream_t stream) {
    (void)in_sizes; (void)n_in; (void)out_size; (void)ws_size;

    const float* x         = (const float*)d_in[0];   // [1024,256]
    const float* pos       = (const float*)d_in[1];   // [1024,3]
    const float* x_skip2   = (const float*)d_in[3];   // [4096,128]
    const float* pos_skip2 = (const float*)d_in[4];   // [4096,3]
    const float* x_skip1   = (const float*)d_in[6];   // [16384,64]
    const float* pos_skip1 = (const float*)d_in[7];   // [16384,3]
    const float* x_skip0   = (const float*)d_in[9];   // [65536,3]
    const float* pos_skip0 = (const float*)d_in[10];  // [65536,3]
    const float* W3a = (const float*)d_in[12]; const float* b3a = (const float*)d_in[13];
    const float* W3b = (const float*)d_in[14]; const float* b3b = (const float*)d_in[15];
    const float* W2a = (const float*)d_in[16]; const float* b2a = (const float*)d_in[17];
    const float* W2b = (const float*)d_in[18]; const float* b2b = (const float*)d_in[19];
    const float* W1a = (const float*)d_in[20]; const float* b1a = (const float*)d_in[21];
    const float* W1b = (const float*)d_in[22]; const float* b1b = (const float*)d_in[23];
    const float* W1c = (const float*)d_in[24]; const float* b1c = (const float*)d_in[25];
    float* out = (float*)d_out;

    // Workspace arena (floats), regions reused across stages. Peak ~51 MB,
    // fully L2-resident on MI455X (192 MB L2).
    float* ws = (float*)d_ws;
    float* bufA = ws;                       // Hcat:  max 65536*67 = 4390912
    float* bufB = ws + 4390912;             // hidden A: max 65536*64
    float* bufC = ws + 4390912 + 4194304;   // hidden B: max 65536*64

    // ---- propagate3: k=4, ns=64, nt=256 per graph, C=256, concat 128 -> 384
    knn_interp_kernel<4><<<dim3(1, 16), 256, 0, stream>>>(
        x, 256, pos, pos_skip2, bufA, 384, 64, 256);
    copy_cols_kernel<<<(4096L * 128 + 255) / 256, 256, 0, stream>>>(
        x_skip2, 128, bufA, 384, 256, 4096L * 128);
    gemm_wmma_kernel<<<dim3(128 / 64, 4096 / 32), 32, 0, stream>>>(
        bufA, W3a, b3a, bufB, 4096, 128, 384, 1);
    gemm_wmma_kernel<<<dim3(128 / 64, 4096 / 32), 32, 0, stream>>>(
        bufB, W3b, b3b, bufC, 4096, 128, 128, 0);

    // ---- propagate2: k=8, ns=256, nt=1024 per graph, C=128, concat 64 -> 192
    knn_interp_kernel<8><<<dim3(4, 16), 256, 0, stream>>>(
        bufC, 128, pos_skip2, pos_skip1, bufA, 192, 256, 1024);
    copy_cols_kernel<<<(16384L * 64 + 255) / 256, 256, 0, stream>>>(
        x_skip1, 64, bufA, 192, 128, 16384L * 64);
    gemm_wmma_kernel<<<dim3(64 / 64, 16384 / 32), 32, 0, stream>>>(
        bufA, W2a, b2a, bufB, 16384, 64, 192, 1);
    gemm_wmma_kernel<<<dim3(64 / 64, 16384 / 32), 32, 0, stream>>>(
        bufB, W2b, b2b, bufC, 16384, 64, 64, 0);

    // ---- propagate1: k=16, ns=1024, nt=4096 per graph, C=64, concat 3 -> 67
    knn_interp_kernel<16><<<dim3(16, 16), 256, 0, stream>>>(
        bufC, 64, pos_skip1, pos_skip0, bufA, 67, 1024, 4096);
    copy_cols_kernel<<<(65536L * 3 + 255) / 256, 256, 0, stream>>>(
        x_skip0, 3, bufA, 67, 64, 65536L * 3);
    gemm_wmma_kernel<<<dim3(64 / 64, 65536 / 32), 32, 0, stream>>>(
        bufA, W1a, b1a, bufB, 65536, 64, 67, 1);
    gemm_wmma_kernel<<<dim3(64 / 64, 65536 / 32), 32, 0, stream>>>(
        bufB, W1b, b1b, bufC, 65536, 64, 64, 1);
    final_linear_kernel<<<65536 / 256, 256, 0, stream>>>(
        bufC, W1c, b1c, out, 65536);
}
